// HybridCardiacODE_27092653703396
// MI455X (gfx1250) — matrix operand order repaired and checked by reading, not compiled, not run on hardware
//
#include <hip/hip_runtime.h>
#include <hip/hip_bf16.h>

typedef __attribute__((ext_vector_type(16))) _Float16 v16h;
typedef __attribute__((ext_vector_type(8)))  _Float16 v8h;
typedef __attribute__((ext_vector_type(4)))  _Float16 v4h;
typedef __attribute__((ext_vector_type(8)))  float    v8f;
typedef __attribute__((ext_vector_type(4)))  float    f32x4;

#define HDIM   512
#define MTILE  128          // rows per workgroup (8 waves x 16 rows)
#define TPB    256
#define LD1    520          // padded f16 row stride for h1 in LDS
#define LDB    40           // padded f16 row stride for staged B slab
#define NGROUPS 4           // 4 groups x 128 columns = 512
#define NT      8           // 16-wide N tiles per group
#define KSTEPS  16          // 512 / 32

// LDS layout (bytes):
//  sW1 10240 | sb1 2048 | sb2 2048 | sW3 4096 | sb3 16 | sV 2560 | snn 1024
//  sh1 128*520*2 = 133120 | sB ping-pong 2 * 128*40*2 = 20480  -> total 175632
static constexpr size_t SMEM_BYTES =
    (5 * HDIM + HDIM + HDIM + 2 * HDIM + 4 + MTILE * 5 + MTILE * 2) * 4
    + (size_t)(MTILE * LD1 + 2 * MTILE * LDB) * 2;

__device__ __forceinline__ float fast_tanh(float x) {
#if __has_builtin(__builtin_amdgcn_tanhf)
  return __builtin_amdgcn_tanhf(x);            // v_tanh_f32 (gfx1250 trans op)
#elif __has_builtin(__builtin_amdgcn_tanh_f32)
  return __builtin_amdgcn_tanh_f32(x);
#else
  // tanh(x) = 1 - 2/(exp(2x)+1); exact limits at +/-inf; v_exp_f32 + v_rcp_f32
  float e = __expf(2.0f * x);
  return 1.0f - 2.0f * __builtin_amdgcn_rcpf(e + 1.0f);
#endif
}

// One-time prep: W2T[n][k] = (f16)W2[k][n], so WMMA B-fragments are K-contiguous.
__global__ void w2_transpose_f16(const float* __restrict__ W2,
                                 _Float16* __restrict__ W2T) {
  int idx = blockIdx.x * blockDim.x + threadIdx.x;   // 0 .. 512*512-1
  int k = idx >> 9;
  int n = idx & (HDIM - 1);
  W2T[(size_t)n * HDIM + k] = (_Float16)W2[idx];
}

__global__ __launch_bounds__(TPB, 1)
void cardiac_fused(const float* __restrict__ t,
                   const float* __restrict__ V,
                   const float* __restrict__ W1, const float* __restrict__ b1,
                   const _Float16* __restrict__ W2T,
                   const float* __restrict__ b2,
                   const float* __restrict__ W3, const float* __restrict__ b3,
                   float* __restrict__ dV) {
  extern __shared__ char smem[];
  float*    sW1 = (float*)smem;                   // 5*512
  float*    sb1 = sW1 + 5 * HDIM;                 // 512
  float*    sb2 = sb1 + HDIM;                     // 512
  float*    sW3 = sb2 + HDIM;                     // 512*2
  float*    sb3 = sW3 + 2 * HDIM;                 // 2 (+pad)
  float*    sV  = sb3 + 4;                        // 128*5
  float*    snn = sV + MTILE * 5;                 // 128*2
  _Float16* sh1 = (_Float16*)(snn + MTILE * 2);   // 128*LD1 (16B aligned)
  _Float16* sB0 = sh1 + MTILE * LD1;              // 128*LDB, buffer 0
  _Float16* sB1 = sB0 + MTILE * LDB;              // 128*LDB, buffer 1

  const int tid     = threadIdx.x;
  const int rowBase = blockIdx.x * MTILE;

  // ---------- cooperative preload ----------
  for (int i = tid; i < 5 * HDIM; i += TPB) sW1[i] = W1[i];
  for (int i = tid; i < HDIM; i += TPB) { sb1[i] = b1[i]; sb2[i] = b2[i]; }
  for (int i = tid; i < 2 * HDIM; i += TPB) sW3[i] = W3[i];
  if (tid < 2) sb3[tid] = b3[tid];
  for (int i = tid; i < MTILE * 5; i += TPB) sV[i] = V[(size_t)rowBase * 5 + i];
  __syncthreads();

  // ---------- phase 1: h1 = tanh(V@W1 + b1) -> f16 LDS (4-wide) ----------
  {
    const int m  = tid >> 1;            // 0..127
    const int kh = (tid & 1) * 256;     // which 256-wide half of k
    const float v0 = sV[m * 5 + 0], v1 = sV[m * 5 + 1], v2 = sV[m * 5 + 2],
                v3 = sV[m * 5 + 3], v4 = sV[m * 5 + 4];
    for (int kk = 0; kk < 256; kk += 4) {
      const int k = kh + kk;
      f32x4 a  = *(const f32x4*)(sb1 + k);
      f32x4 w0 = *(const f32x4*)(sW1 + 0 * HDIM + k);
      f32x4 w1 = *(const f32x4*)(sW1 + 1 * HDIM + k);
      f32x4 w2 = *(const f32x4*)(sW1 + 2 * HDIM + k);
      f32x4 w3 = *(const f32x4*)(sW1 + 3 * HDIM + k);
      f32x4 w4 = *(const f32x4*)(sW1 + 4 * HDIM + k);
      a = a + v0 * w0 + v1 * w1 + v2 * w2 + v3 * w3 + v4 * w4;
      v4h out;
      #pragma unroll
      for (int e = 0; e < 4; ++e) out[e] = (_Float16)fast_tanh(a[e]);
      *(v4h*)(sh1 + m * LD1 + k) = out;
    }
  }
  __syncthreads();

  // ---------- phase 2: h2 GEMM via WMMA, fused tanh + W3 epilogue ----------
  const int wave  = tid >> 5;
  const int lane  = tid & 31;
  const int lhalf = lane >> 4;          // 0: M rows 0-7 / 1: M rows 8-15 of tile
  const int l15   = lane & 15;
  const int mWave = wave * 16;

  float p0[8], p1[8];                   // per-lane partial nn_out accumulators
  #pragma unroll
  for (int j = 0; j < 8; ++j) { p0[j] = 0.0f; p1[j] = 0.0f; }

  const int nLocal = tid >> 1;          // B-staging: row within 128-col group
  const int kOff   = (tid & 1) * 16;    // 16 f16 per thread per half-row

  for (int ng = 0; ng < NGROUPS; ++ng) {
    const int nBase = ng * 128;
    v8f acc[NT] = {};

    const _Float16* gBase = W2T + (size_t)(nBase + nLocal) * HDIM + kOff;
    _Float16* stDst = sB0 + nLocal * LDB + kOff;   // this thread's slot, buf 0
    const int stFlip = MTILE * LDB;                // buf0 <-> buf1 offset

    // prologue: slab kt=0 into buf0, prefetch kt=1 into regs
    f32x4 r0 = *(const f32x4*)(gBase + 0);
    f32x4 r1 = *(const f32x4*)(gBase + 8);
    *(f32x4*)(stDst + 0) = r0;
    *(f32x4*)(stDst + 8) = r1;
    r0 = *(const f32x4*)(gBase + 32);
    r1 = *(const f32x4*)(gBase + 40);
    __syncthreads();

    for (int kt = 0; kt < KSTEPS; ++kt) {
      // stage slab kt+1 into the buffer not being read this step
      if (kt + 1 < KSTEPS) {
        _Float16* w = stDst + ((kt + 1) & 1) * stFlip;
        *(f32x4*)(w + 0) = r0;
        *(f32x4*)(w + 8) = r1;
        if (kt + 2 < KSTEPS) {
          const _Float16* gN = gBase + (kt + 2) * 32;
          r0 = *(const f32x4*)(gN + 0);
          r1 = *(const f32x4*)(gN + 8);
        }
      }
      const _Float16* bufR = (kt & 1) ? sB1 : sB0;

      // A fragment (16x32 f16): lanes 0-15 K=0-7 & 16-23, lanes 16-31 +8
      const _Float16* aPtr = sh1 + (mWave + l15) * LD1 + kt * 32 + lhalf * 8;
      v8h a0 = *(const v8h*)(aPtr);
      v8h a1 = *(const v8h*)(aPtr + 16);
      v16h afrag;
      #pragma unroll
      for (int e = 0; e < 8; ++e) { afrag[e] = a0[e]; afrag[8 + e] = a1[e]; }

      #pragma unroll
      for (int nt = 0; nt < NT; ++nt) {
        // B fragment (32x16 f16): lane n = l15, lanes 0-15 K=0-7 & 16-23, lanes 16-31 +8
        const _Float16* bPtr = bufR + (nt * 16 + l15) * LDB + lhalf * 8;
        v8h bb0 = *(const v8h*)(bPtr);
        v8h bb1 = *(const v8h*)(bPtr + 16);
        v16h bfrag;
        #pragma unroll
        for (int e = 0; e < 8; ++e) { bfrag[e] = bb0[e]; bfrag[8 + e] = bb1[e]; }
        acc[nt] = __builtin_amdgcn_wmma_f32_16x16x32_f16(
            false, afrag, false, bfrag, (short)0, acc[nt], false, false);
      }
      __syncthreads();    // one barrier per K-step (ping-pong staging)
    }

    // fused epilogue: +b2, tanh, accumulate the two W3 columns
    #pragma unroll
    for (int nt = 0; nt < NT; ++nt) {
      const int   n    = nBase + nt * 16 + l15;
      const float bias = sb2[n];
      const float w30  = sW3[n * 2 + 0];
      const float w31  = sW3[n * 2 + 1];
      #pragma unroll
      for (int j = 0; j < 8; ++j) {     // C layout: VGPR j <-> M = j + lhalf*8
        const float h = fast_tanh(acc[nt][j] + bias);
        p0[j] = fmaf(h, w30, p0[j]);
        p1[j] = fmaf(h, w31, p1[j]);
      }
    }
  }

  // butterfly-reduce partials over the 16 lanes of each half-wave
  #pragma unroll
  for (int j = 0; j < 8; ++j) {
    #pragma unroll
    for (int off = 1; off < 16; off <<= 1) {
      p0[j] += __shfl_xor(p0[j], off, 32);
      p1[j] += __shfl_xor(p1[j], off, 32);
    }
  }
  if (l15 == 0) {
    #pragma unroll
    for (int j = 0; j < 8; ++j) {
      const int m = mWave + lhalf * 8 + j;
      snn[m * 2 + 0] = p0[j] + sb3[0];
      snn[m * 2 + 1] = p1[j] + sb3[1];
    }
  }
  __syncthreads();

  // ---------- phase 3: cardiac ODE + store dV ----------
  if (tid < MTILE) {
    const int m = tid;
    const int s = rowBase + m;
    float tm = t[s];
    tm = tm - 0.8f * floorf(tm * 1.25f);
    float ev = 0.0f, ea = 0.0f;
    if (tm < 0.3f)  { float x = __sinf(3.14159265358979f * tm * (1.0f / 0.3f));  ev = x * x; }
    if (tm < 0.17f) { float x = __sinf(3.14159265358979f * tm * (1.0f / 0.17f)); ea = x * x; }
    const float E_LV = 0.10f + 2.40f * ev;
    const float E_RV = 0.05f + 0.55f * ev;
    const float E_LA = 0.15f + 0.10f * ea;
    const float E_RA = 0.10f + 0.10f * ea;
    const float V_LA = sV[m * 5 + 0], V_LV = sV[m * 5 + 1], V_Ao = sV[m * 5 + 2],
                V_RA = sV[m * 5 + 3], V_RV = sV[m * 5 + 4];
    const float P_peri = snn[m * 2 + 0];
    const float V_spt  = snn[m * 2 + 1];
    const float P_LV = E_LV * (V_LV - V_spt - 10.0f) + P_peri;
    const float P_RV = E_RV * (V_RV + V_spt - 10.0f) + P_peri;
    const float P_LA = E_LA * (V_LA - 5.0f);
    const float P_RA = E_RA * (V_RA - 5.0f);
    const float P_Ao = (V_Ao - 50.0f) * (1.0f / 1.5f);
    const float Q_mit = fmaxf((P_LA - P_LV) * 100.0f, 0.0f);
    const float Q_ao  = fmaxf((P_LV - P_Ao) * 100.0f, 0.0f);
    const float Q_tri = fmaxf((P_RA - P_RV) * 100.0f, 0.0f);
    const float Q_pul = fmaxf((P_RV - 15.0f) * 20.0f, 0.0f);
    const float Q_sys = P_Ao - 5.0f;
    const float Q_vc  = (5.0f - P_RA) * 20.0f;
    const float Q_pv  = (15.0f - P_LA) * 20.0f;
    float* o = dV + (size_t)s * 5;
    o[0] = Q_pv - Q_mit;
    o[1] = Q_mit - Q_ao;
    o[2] = Q_ao - Q_sys;
    o[3] = Q_vc - Q_tri;
    o[4] = Q_tri - Q_pul;
  }
}

extern "C" void kernel_launch(void* const* d_in, const int* in_sizes, int n_in,
                              void* d_out, int out_size, void* d_ws, size_t ws_size,
                              hipStream_t stream) {
  const float* t  = (const float*)d_in[0];
  const float* V  = (const float*)d_in[1];
  const float* W1 = (const float*)d_in[2];
  const float* b1 = (const float*)d_in[3];
  const float* W2 = (const float*)d_in[4];
  const float* b2 = (const float*)d_in[5];
  const float* W3 = (const float*)d_in[6];
  const float* b3 = (const float*)d_in[7];
  float*    dV  = (float*)d_out;
  _Float16* W2T = (_Float16*)d_ws;     // 512 KB scratch, rewritten every call

  const int B = in_sizes[0];

  // allow >64KB dynamic LDS (gfx1250 WGP has 320KB); deterministic host call
  (void)hipFuncSetAttribute((const void*)cardiac_fused,
                            hipFuncAttributeMaxDynamicSharedMemorySize,
                            (int)SMEM_BYTES);

  w2_transpose_f16<<<(HDIM * HDIM) / TPB, TPB, 0, stream>>>(W2, W2T);
  cardiac_fused<<<B / MTILE, TPB, SMEM_BYTES, stream>>>(
      t, V, W1, b1, W2T, b2, W3, b3, dV);
}
